// RFC_block_30794915512712
// MI455X (gfx1250) — matrix operand order repaired and checked by reading, not compiled, not run on hardware
//
#include <hip/hip_runtime.h>
#include <stdint.h>

// ---------------------------------------------------------------------------
// Spiking recurrent net: ff = x@W_fc^T + b_fc once, then 16 dependent steps of
//   mem_r = spk@(W_rec*mask)^T + b_rec  (f16 WMMA GEMM, f32 accumulate)
//   + fused elementwise spiking-state update in the GEMM epilogue.
// Inner loop: double-buffered LDS, async global->LDS copies (ASYNCcnt),
// 8 v_wmma_f32_16x16x32_f16 per wave per K-step.
// ---------------------------------------------------------------------------

typedef __attribute__((ext_vector_type(16))) _Float16     v16h;
typedef __attribute__((ext_vector_type(8)))  float        v8f;
typedef __attribute__((ext_vector_type(8)))  unsigned int v8u;

#define THRESH     0.5f
#define DECAY      0.2f
#define XISHUSPIKE 0.5f
#define XISHUADA   0.2f

// CDNA5 async global->LDS path (ASYNCcnt). Guarded so the build stays green
// on toolchains without the builtin (and on the host pass).
#if defined(__HIP_DEVICE_COMPILE__) && defined(__has_builtin)
#if __has_builtin(__builtin_amdgcn_global_load_async_to_lds_b128) && \
    __has_builtin(__builtin_amdgcn_s_wait_asynccnt)
#define USE_ASYNC_LDS 1
#endif
#endif
#ifndef USE_ASYNC_LDS
#define USE_ASYNC_LDS 0
#endif

#if USE_ASYNC_LDS
// Builtin signature (from hipcc diagnostic): param 0 is
// 'int __attribute__((vector_size(16))) AS(1) *', so match it exactly.
typedef int v4i __attribute__((vector_size(4 * sizeof(int))));
typedef __attribute__((address_space(1))) v4i as1_v4i;
typedef __attribute__((address_space(3))) v4i as3_v4i;
#endif

// Copy one 16-byte chunk global->LDS (async if available).
__device__ __forceinline__ void cp16(const _Float16* g, _Float16* l) {
#if USE_ASYNC_LDS
  __builtin_amdgcn_global_load_async_to_lds_b128(
      (as1_v4i*)(uintptr_t)g, (as3_v4i*)(uint32_t)(uintptr_t)l, 0, 0);
#else
  *(uint4*)l = *(const uint4*)g;
#endif
}

// Drain this wave's async copies, then rendezvous the workgroup.
__device__ __forceinline__ void async_join() {
#if USE_ASYNC_LDS
  __builtin_amdgcn_s_wait_asynccnt(0);
#endif
  __syncthreads();
}

// Cast weights to f16 (recurrent weight masked). IN == OUT so one pass.
__global__ void snn_prep_weights(const float* __restrict__ Wfc,
                                 const float* __restrict__ Wrec,
                                 const int*   __restrict__ mask,
                                 _Float16* __restrict__ Wfch,
                                 _Float16* __restrict__ Wrh, int n) {
  int i = blockIdx.x * blockDim.x + threadIdx.x;
  if (i < n) {
    Wfch[i] = (_Float16)Wfc[i];
    Wrh[i]  = (_Float16)(mask[i] ? Wrec[i] : 0.0f);
  }
}

// Cast x to f16 and zero all state (mem, ath, ssum=d_out, spk ping buffer).
__global__ void snn_init(const float* __restrict__ x,
                         _Float16* __restrict__ xh,
                         float* __restrict__ mem, float* __restrict__ ath,
                         float* __restrict__ ssum, _Float16* __restrict__ spk0,
                         int n) {
  int i = blockIdx.x * blockDim.x + threadIdx.x;
  if (i < n) {
    xh[i]   = (_Float16)x[i];
    mem[i]  = 0.0f;
    ath[i]  = 0.0f;
    ssum[i] = 0.0f;
    spk0[i] = (_Float16)0.0f;
  }
}

// C[M,N] = A[M,K] * B[N,K]^T + bias, f16 in / f32 accum.
// mode 0: store raw result (ff pass).  mode 1: fused spiking update.
// Block = 128 threads (4 wave32s, 2x2 wave grid); block tile 64(M) x 128(N);
// each wave owns 32x64 (2 M-frags x 4 N-frags = 8 WMMAs per K-step of 32).
__launch_bounds__(128, 2)
__global__ void snn_gemm(const _Float16* __restrict__ A,    // [M,K] row-major
                         const _Float16* __restrict__ Bm,   // [N,K] row-major
                         const float*    __restrict__ bias, // [N]
                         float*          __restrict__ ffout,// mode 0 dest
                         const float*    __restrict__ ff,   // mode 1 src
                         float* __restrict__ mem,
                         float* __restrict__ ath,
                         float* __restrict__ ssum,
                         const _Float16* __restrict__ spk_cur,
                         _Float16*       __restrict__ spk_next,
                         int M, int N, int K, int mode) {
  constexpr int TM = 64, TN = 128, TK = 32;
  constexpr int LD = TK + 8;                  // 40 halfs = 80 B (16-B multiple)
  __shared__ __align__(16) _Float16 As[2][TM][LD];   // double buffered
  __shared__ __align__(16) _Float16 Bs[2][TN][LD];

  const int tid  = threadIdx.x;
  const int wave = tid >> 5;
  const int lane = tid & 31;
  const int wm   = wave >> 1;                 // 0..1 : 32-row slice of tile
  const int wn   = wave & 1;                  // 0..1 : 64-col slice of tile
  const long m0  = (long)blockIdx.y * TM;
  const long n0  = (long)blockIdx.x * TN;

  v8f zero = {0.f, 0.f, 0.f, 0.f, 0.f, 0.f, 0.f, 0.f};
  v8f acc[2][4] = {{zero, zero, zero, zero}, {zero, zero, zero, zero}};

  // Stage one (A 64x32, B 128x32) f16 K-tile into LDS buffer `buf`.
  auto stage = [&](int buf, long k0) {
#pragma unroll
    for (int c = 0; c < 2; ++c) {             // A: 256 chunks / 128 threads
      const int idx = tid + 128 * c;
      const int row = idx >> 2, seg = idx & 3;
      cp16(A + (m0 + row) * (long)K + k0 + seg * 8, &As[buf][row][seg * 8]);
    }
#pragma unroll
    for (int c = 0; c < 4; ++c) {             // B: 512 chunks / 128 threads
      const int idx = tid + 128 * c;
      const int row = idx >> 2, seg = idx & 3;
      cp16(Bm + (n0 + row) * (long)K + k0 + seg * 8, &Bs[buf][row][seg * 8]);
    }
  };

  stage(0, 0);
  async_join();

  for (int k0 = 0; k0 < K; k0 += TK) {
    const int cur = (k0 / TK) & 1;
    if (k0 + TK < K) stage(cur ^ 1, k0 + TK); // overlap next fill with compute

    // A fragments (16x32 f16): lanes 0-15 -> K {0..7,16..23},
    // lanes 16-31 -> K {8..15,24..31}; two ds_load_b128 per fragment.
    const int kb = (lane >> 4) * 8;
    v16h af[2];
#pragma unroll
    for (int mi = 0; mi < 2; ++mi) {
      const int fr = wm * 32 + mi * 16 + (lane & 15);
      uint4 a0 = *(const uint4*)&As[cur][fr][kb];
      uint4 a1 = *(const uint4*)&As[cur][fr][kb + 16];
      v8u  au  = {a0.x, a0.y, a0.z, a0.w, a1.x, a1.y, a1.z, a1.w};
      af[mi]   = __builtin_bit_cast(v16h, au);
    }

    // B fragments (32x16 f16): lanes 0-15 K {0..7,16..23}, lanes 16-31 K {8..15,24..31}.
    const int kb2 = (lane >> 4) * 8;
#pragma unroll
    for (int nt = 0; nt < 4; ++nt) {
      const int br = wn * 64 + nt * 16 + (lane & 15);
      uint4 b0 = *(const uint4*)&Bs[cur][br][kb2];
      uint4 b1 = *(const uint4*)&Bs[cur][br][kb2 + 16];
      v8u  bu  = {b0.x, b0.y, b0.z, b0.w, b1.x, b1.y, b1.z, b1.w};
      v16h bf  = __builtin_bit_cast(v16h, bu);
#pragma unroll
      for (int mi = 0; mi < 2; ++mi)
        acc[mi][nt] = __builtin_amdgcn_wmma_f32_16x16x32_f16(
            false, af[mi], false, bf, (short)0, acc[mi][nt], false, false);
    }
    async_join();                              // next buffer ready for all waves
  }

  // Epilogue: C/D layout -> VGPR r holds row r + 8*(lane>>4), col lane&15.
  const int cn   = lane & 15;
  const int rsel = lane >> 4;
#pragma unroll
  for (int mi = 0; mi < 2; ++mi) {
#pragma unroll
    for (int nt = 0; nt < 4; ++nt) {
      const long gn = n0 + wn * 64 + nt * 16 + cn;
      const float bv = bias[gn];
#pragma unroll
      for (int r = 0; r < 8; ++r) {
        const long gm  = m0 + wm * 32 + mi * 16 + rsel * 8 + r;
        const long g   = gm * (long)N + gn;
        const float val = acc[mi][nt][r] + bv;
        if (mode == 0) {
          ffout[g] = val;                      // ff = x@W_fc^T + b_fc
        } else {
          // fused spiking update: val == mem_r at (gm, gn)
          const float s_old  = (float)spk_cur[g];          // exact in f16
          const float a_new  = 0.9f * ath[g] + XISHUSPIKE * s_old;
          const float mem_ff = mem[g] * DECAY * (1.0f - s_old)
                               + ff[g] - a_new * XISHUADA;
          const float s_ff = (mem_ff > THRESH) ? 1.0f : 0.0f;
          const float s_r  = (val    > THRESH) ? 1.0f : 0.0f;
          mem[g] = mem_ff + val;
          ath[g] = a_new;
          const float s_new = s_ff + s_r;
          ssum[g] += s_new;
          spk_next[g] = (_Float16)s_new;       // ping-pong: no intra-launch race
        }
      }
    }
  }
}

extern "C" void kernel_launch(void* const* d_in, const int* in_sizes, int n_in,
                              void* d_out, int out_size, void* d_ws, size_t ws_size,
                              hipStream_t stream) {
  (void)in_sizes; (void)n_in; (void)out_size; (void)ws_size;
  const int Bdim = 512, IN = 2048, OUT = 2048, T = 16;

  const float* x     = (const float*)d_in[0];
  const float* W_fc  = (const float*)d_in[1];
  const float* b_fc  = (const float*)d_in[2];
  const float* W_rec = (const float*)d_in[3];
  const float* b_rec = (const float*)d_in[4];
  const int*   mask  = (const int*)d_in[5];
  float* ssum = (float*)d_out;

  // carve 256-B aligned scratch regions out of d_ws (~34 MB total)
  char* ws = (char*)d_ws;
  size_t off = 0;
  auto carve = [&](size_t bytes) -> void* {
    void* p = ws + off;
    off = (off + bytes + 255) & ~(size_t)255;
    return p;
  };
  _Float16* Wfch = (_Float16*)carve((size_t)OUT * IN * 2);
  _Float16* Wrh  = (_Float16*)carve((size_t)OUT * OUT * 2);
  _Float16* xh   = (_Float16*)carve((size_t)Bdim * IN * 2);
  float*    ffb  = (float*)   carve((size_t)Bdim * OUT * 4);
  float*    mem  = (float*)   carve((size_t)Bdim * OUT * 4);
  float*    ath  = (float*)   carve((size_t)Bdim * OUT * 4);
  _Float16* spk0 = (_Float16*)carve((size_t)Bdim * OUT * 2);
  _Float16* spk1 = (_Float16*)carve((size_t)Bdim * OUT * 2);

  const int nw = OUT * IN;
  snn_prep_weights<<<(nw + 255) / 256, 256, 0, stream>>>(W_fc, W_rec, mask,
                                                         Wfch, Wrh, nw);
  const int ns = Bdim * OUT;
  snn_init<<<(ns + 255) / 256, 256, 0, stream>>>(x, xh, mem, ath, ssum, spk0, ns);

  dim3 grid(OUT / 128, Bdim / 64), block(128);
  // ff = x @ W_fc^T + b_fc  (mode 0)
  snn_gemm<<<grid, block, 0, stream>>>(xh, Wfch, b_fc, ffb,
                                       nullptr, nullptr, nullptr, nullptr,
                                       nullptr, nullptr,
                                       Bdim, OUT, IN, 0);
  // 16 dependent recurrent steps, spike buffers ping-pong (mode 1)
  for (int t = 0; t < T; ++t) {
    _Float16* sc = (t & 1) ? spk1 : spk0;
    _Float16* sn = (t & 1) ? spk0 : spk1;
    snn_gemm<<<grid, block, 0, stream>>>(sc, Wrh, b_rec, nullptr,
                                         ffb, mem, ath, ssum, sc, sn,
                                         Bdim, OUT, OUT, 1);
  }
}
